// S3Tokenizer_24275155157654
// MI455X (gfx1250) — compile-verified, matrix operand order checked by reading, not run on hardware
//
#include <hip/hip_runtime.h>
#include <math.h>

// ---------------------------------------------------------------- constants
#define NMELS_  128
#define D_      1280
#define NHEAD_  20
#define NLAYER_ 6
#define NCODE_  4096
#define BATCH_  8
#define TIN_    3000
#define T1_     1500
#define TOK_    750
#define MTOT_   (BATCH_*TOK_)   // 6000
#define MPAD_   6016            // MTOT_ rounded to 128 (async staging over-read pad)
#define HD_     64
#define FF_     5120
#define K2_     (D_*3)          // 3840 im2col K for conv2
#define NEGBIG  (-1.0e10f)

typedef __attribute__((ext_vector_type(16))) __bf16 v16bf;
typedef __attribute__((ext_vector_type(8)))  float  v8f;
typedef __attribute__((ext_vector_type(4)))  int    v4i;

#define LDST 40   // padded LDS row stride in bf16 elems (80B, 16B aligned, conflict-free)

// -------- CDNA5 async global->LDS path (ASYNCcnt), guarded for portability
#if defined(__has_builtin)
#if __has_builtin(__builtin_amdgcn_global_load_async_to_lds_b128)
#define HAVE_ASYNC_LDS 1
#endif
#endif
#ifndef HAVE_ASYNC_LDS
#define HAVE_ASYNC_LDS 0
#endif

#if HAVE_ASYNC_LDS
typedef __attribute__((address_space(1))) v4i gv4i_t;   // global int4
typedef __attribute__((address_space(3))) v4i lv4i_t;   // LDS int4
__device__ __forceinline__ void async_b128(const void* g, void* l) {
    __builtin_amdgcn_global_load_async_to_lds_b128(
        (gv4i_t*)(g), (lv4i_t*)(l), 0, 0);
}
#if __has_builtin(__builtin_amdgcn_s_wait_asynccnt)
#define WAIT_ASYNC(n) __builtin_amdgcn_s_wait_asynccnt(n)
#else
#define WAIT_ASYNC(n) asm volatile("s_wait_asynccnt %0" :: "i"(n) : "memory")
#endif
#endif

// ---------------------------------------------------------------- device utils
__device__ __forceinline__ float gelu_f(float x) {
    return 0.5f * x * (1.0f + erff(x * 0.70710678118654752f));
}

__device__ __forceinline__ float block_sum(float v, float* sb) {
    int lane = threadIdx.x & 31, w = threadIdx.x >> 5;
#pragma unroll
    for (int o = 16; o; o >>= 1) v += __shfl_xor(v, o, 32);
    if (lane == 0) sb[w] = v;
    __syncthreads();
    float r;
    if (threadIdx.x == 0) {
        r = 0.f;
        for (int i = 0; i < 8; i++) r += sb[i];
        sb[0] = r;
    }
    __syncthreads();
    r = sb[0];
    __syncthreads();
    return r;
}

union FragU { uint4 u[2]; v16bf v; };

// Load a 16x32 bf16 fragment per ISA 7.12.2: lane m = lane&15, k-half = lane>>4.
// elements [0..7]  <- k in [half*8 .. half*8+7]       (contiguous 16B)
// elements [8..15] <- k in [16+half*8 .. 16+half*8+7] (contiguous 16B)
__device__ __forceinline__ v16bf load_frag(const __bf16* rowbase, int half) {
    FragU f;
    const __bf16* p = rowbase + half * 8;
    f.u[0] = *(const uint4*)(p);
    f.u[1] = *(const uint4*)(p + 16);
    return f.v;
}

// ---------------------------------------------------------------- WMMA GEMM
// C[M,N] = epilogue( A[M,K](bf16) x Bw[N,K]^T(bf16) )
// epilogue: r = (acc + bias[n]) * scale - colsub[n]; if(gelu) r = gelu(r);
//           if(residual) r += residual[m,n]; store to outF (f32) and/or outH (bf16)
// NOTE (async path): A rows in [M, gridDim.y*128) are read from workspace pad; their
// products land only in unstored C rows, so no zero-fill is needed.
__global__ __launch_bounds__(256)
void gemm_bf16_wmma(const __bf16* __restrict__ A, const __bf16* __restrict__ Bw,
                    const float* __restrict__ bias, const float* __restrict__ colsub,
                    const float* __restrict__ residual, float scale,
                    int M, int N, int K, int dogelu,
                    float* __restrict__ outF, __bf16* __restrict__ outH) {
    __shared__ __align__(16) __bf16 sA[2][128 * LDST];
    __shared__ __align__(16) __bf16 sB[2][128 * LDST];

    const int tid  = threadIdx.x;
    const int lane = tid & 31;
    const int wave = tid >> 5;       // 0..7
    const int wm   = wave & 3;       // 4 waves along M, 32 rows each
    const int wn   = wave >> 2;      // 2 waves along N, 64 cols each
    const int mBase = blockIdx.y * 128;
    const int nBase = blockIdx.x * 128;
    const int rlan = lane & 15;
    const int half = lane >> 4;

    v8f acc[2][4] = {};
    const int nsteps = K / 32;

#if HAVE_ASYNC_LDS
    // per-thread staging slots: 2x 16B of A and 2x 16B of B per tile
    int srow[2], sseg[2];
#pragma unroll
    for (int i = 0; i < 2; i++) {
        int li = tid + i * 256;          // 0..511
        srow[i] = li >> 2;               // 0..127
        sseg[i] = li & 3;                // 4x 16B per 32-elem row
    }
    // prologue: stage tile 0 into buffer 0 (4 async ops / thread)
#pragma unroll
    for (int i = 0; i < 2; i++) {
        async_b128(A  + (size_t)(mBase + srow[i]) * K + sseg[i] * 8,
                   &sA[0][srow[i] * LDST + sseg[i] * 8]);
        async_b128(Bw + (size_t)(nBase + srow[i]) * K + sseg[i] * 8,
                   &sB[0][srow[i] * LDST + sseg[i] * 8]);
    }
    for (int s = 0; s < nsteps; s++) {
        const int cur = s & 1, nxt = cur ^ 1;
        const bool more = (s + 1) < nsteps;
        if (more) {
            const int k1 = (s + 1) * 32;
#pragma unroll
            for (int i = 0; i < 2; i++) {
                async_b128(A  + (size_t)(mBase + srow[i]) * K + k1 + sseg[i] * 8,
                           &sA[nxt][srow[i] * LDST + sseg[i] * 8]);
                async_b128(Bw + (size_t)(nBase + srow[i]) * K + k1 + sseg[i] * 8,
                           &sB[nxt][srow[i] * LDST + sseg[i] * 8]);
            }
            WAIT_ASYNC(4);   // tile `cur` (4 older ops) has landed; `nxt` in flight
        } else {
            WAIT_ASYNC(0);
        }
        __syncthreads();

        v16bf afr[2], bfr[4];
#pragma unroll
        for (int f = 0; f < 2; f++)
            afr[f] = load_frag(&sA[cur][(wm * 32 + f * 16 + rlan) * LDST], half);
#pragma unroll
        for (int f = 0; f < 4; f++)
            bfr[f] = load_frag(&sB[cur][(wn * 64 + f * 16 + rlan) * LDST], half);
#pragma unroll
        for (int i = 0; i < 2; i++)
#pragma unroll
            for (int j = 0; j < 4; j++)
                acc[i][j] = __builtin_amdgcn_wmma_f32_16x16x32_bf16(
                    false, afr[i], false, bfr[j], (short)0, acc[i][j], false, false);
        __syncthreads();   // all waves done reading `cur` before it is re-staged
    }
#else
    // fallback: synchronous staging through VGPRs
    for (int s = 0; s < nsteps; s++) {
        const int k0 = s * 32;
        if (k0 + 32 < K) {
            int prow = tid >> 1;
            __builtin_prefetch(A  + (size_t)(mBase + prow) * K + k0 + 32 + (tid & 1) * 16, 0, 3);
            __builtin_prefetch(Bw + (size_t)(nBase + prow) * K + k0 + 32 + (tid & 1) * 16, 0, 3);
        }
#pragma unroll
        for (int i = 0; i < 2; i++) {
            int li = tid + i * 256;
            int row = li >> 2, seg = li & 3;
            uint4 va = *(const uint4*)(A + (size_t)(mBase + row) * K + k0 + seg * 8);
            *(uint4*)&sA[0][row * LDST + seg * 8] = va;
            uint4 vb = *(const uint4*)(Bw + (size_t)(nBase + row) * K + k0 + seg * 8);
            *(uint4*)&sB[0][row * LDST + seg * 8] = vb;
        }
        __syncthreads();
        v16bf afr[2], bfr[4];
#pragma unroll
        for (int f = 0; f < 2; f++)
            afr[f] = load_frag(&sA[0][(wm * 32 + f * 16 + rlan) * LDST], half);
#pragma unroll
        for (int f = 0; f < 4; f++)
            bfr[f] = load_frag(&sB[0][(wn * 64 + f * 16 + rlan) * LDST], half);
#pragma unroll
        for (int i = 0; i < 2; i++)
#pragma unroll
            for (int j = 0; j < 4; j++)
                acc[i][j] = __builtin_amdgcn_wmma_f32_16x16x32_bf16(
                    false, afr[i], false, bfr[j], (short)0, acc[i][j], false, false);
        __syncthreads();
    }
#endif

    // epilogue; C layout: VGPR e -> M = e + half*8, N = lane&15
#pragma unroll
    for (int i = 0; i < 2; i++) {
        int rowf = mBase + wm * 32 + i * 16;
#pragma unroll
        for (int j = 0; j < 4; j++) {
            int col = nBase + wn * 64 + j * 16 + rlan;
            float bi = bias   ? bias[col]   : 0.f;
            float cs = colsub ? colsub[col] : 0.f;
#pragma unroll
            for (int e = 0; e < 8; e++) {
                int row = rowf + e + half * 8;
                if (row < M && col < N) {
                    float r = (acc[i][j][e] + bi) * scale - cs;
                    if (dogelu) r = gelu_f(r);
                    if (residual) r += residual[(size_t)row * N + col];
                    if (outF) outF[(size_t)row * N + col] = r;
                    if (outH) outH[(size_t)row * N + col] = (__bf16)r;
                }
            }
        }
    }
}

// ---------------------------------------------------------------- small kernels
__global__ void tobf16_k(const float* __restrict__ s, __bf16* __restrict__ d, long n) {
    long i = (long)blockIdx.x * 256 + threadIdx.x;
    if (i < n) d[i] = (__bf16)s[i];
}

__global__ __launch_bounds__(256)
void cb2_k(const float* __restrict__ cb, float* __restrict__ cb2) {
    __shared__ float sb[8];
    int row = blockIdx.x;
    const float* r = cb + (size_t)row * D_;
    float s = 0.f;
    for (int i = threadIdx.x; i < D_; i += 256) { float v = r[i]; s += v * v; }
    s = block_sum(s, sb);
    if (threadIdx.x == 0) cb2[row] = s;
}

// conv1: mel (B,128,3000) -> x1 (B,1500,1280) fp32, GELU, stride 2 pad 1
__global__ __launch_bounds__(256)
void conv1_k(const float* __restrict__ mel, const float* __restrict__ w,
             const float* __restrict__ bias, float* __restrict__ x1) {
    int o = blockIdx.x * 256 + threadIdx.x;  // 0..1279
    int t = blockIdx.y;                      // 0..1499
    int b = blockIdx.z;
    float acc = bias[o];
    const float* wrow = w + (size_t)o * NMELS_ * 3;
    const float* mb   = mel + (size_t)b * NMELS_ * TIN_;
    for (int c = 0; c < NMELS_; c++) {
        const float* mc = mb + (size_t)c * TIN_;
        int u = 2 * t - 1;
        float m0 = (u >= 0) ? mc[u] : 0.f;
        float m1 = mc[u + 1];
        float m2 = mc[u + 2];
        acc += m0 * wrow[c * 3] + m1 * wrow[c * 3 + 1] + m2 * wrow[c * 3 + 2];
    }
    x1[((size_t)b * T1_ + t) * D_ + o] = gelu_f(acc);
}

// im2col for conv2: A2[(b*750+t)*3840 + c*3+k] = x1[b, 2t+k-1, c] (bf16, zero pad)
__global__ void im2col_k(const float* __restrict__ x1, __bf16* __restrict__ A2) {
    size_t idx = (size_t)blockIdx.x * 256 + threadIdx.x;
    if (idx >= (size_t)MTOT_ * K2_) return;
    int col = (int)(idx % K2_);
    size_t rt = idx / K2_;
    int t = (int)(rt % TOK_);
    int b = (int)(rt / TOK_);
    int c = col / 3, k = col % 3;
    int u = 2 * t + k - 1;
    float v = (u >= 0 && u < T1_) ? x1[((size_t)b * T1_ + u) * D_ + c] : 0.f;
    A2[idx] = (__bf16)v;
}

__global__ void posadd_k(float* __restrict__ x, const float* __restrict__ pos) {
    size_t idx = (size_t)blockIdx.x * 256 + threadIdx.x;   // exactly 6000*1280
    int colD = (int)(idx % D_);
    int t = (int)((idx / D_) % TOK_);
    x[idx] += pos[(size_t)t * D_ + colD];
}

__global__ __launch_bounds__(256)
void layernorm_k(const float* __restrict__ x, const float* __restrict__ g,
                 const float* __restrict__ b, __bf16* __restrict__ out) {
    __shared__ float sb[8];
    int row = blockIdx.x;
    const float* xr = x + (size_t)row * D_;
    float s = 0.f, s2 = 0.f;
    for (int i = threadIdx.x; i < D_; i += 256) { float v = xr[i]; s += v; s2 += v * v; }
    s  = block_sum(s,  sb);
    s2 = block_sum(s2, sb);
    float mean = s / D_;
    float var  = s2 / D_ - mean * mean;
    float rstd = rsqrtf(var + 1e-5f);
    __bf16* orow = out + (size_t)row * D_;
    for (int i = threadIdx.x; i < D_; i += 256)
        orow[i] = (__bf16)((xr[i] - mean) * rstd * g[i] + b[i]);
}

// one wave32 per query, online softmax over 750 keys; head dim 64 => 2 elems/lane
__global__ __launch_bounds__(256)
void attention_k(const __bf16* __restrict__ q, const __bf16* __restrict__ kk,
                 const __bf16* __restrict__ v, const int* __restrict__ mel_len,
                 __bf16* __restrict__ out) {
    int wave = threadIdx.x >> 5, lane = threadIdx.x & 31;
    int t = blockIdx.x * 8 + wave;
    if (t >= TOK_) return;
    int h = blockIdx.y, b = blockIdx.z;
    size_t base = ((size_t)(b * TOK_ + t) * D_ + h * HD_);
    float q0 = (float)q[base + lane], q1 = (float)q[base + lane + 32];
    int   ml = mel_len[b];
    float m = -3.4e38f, ssum = 0.f, a0 = 0.f, a1 = 0.f;
    for (int j = 0; j < TOK_; j++) {
        size_t kb = ((size_t)(b * TOK_ + j) * D_ + h * HD_);
        float s = q0 * (float)kk[kb + lane] + q1 * (float)kk[kb + lane + 32];
#pragma unroll
        for (int o = 16; o; o >>= 1) s += __shfl_xor(s, o, 32);
        if (3 + 4 * j >= ml) s += NEGBIG;
        float mn = fmaxf(m, s);
        float corr = __expf(m - mn);
        float p = __expf(s - mn);
        ssum = ssum * corr + p;
        a0 = a0 * corr + p * (float)v[kb + lane];
        a1 = a1 * corr + p * (float)v[kb + lane + 32];
        m = mn;
    }
    float inv = 1.0f / ssum;
    out[base + lane]      = (__bf16)(a0 * inv);
    out[base + lane + 32] = (__bf16)(a1 * inv);
}

__global__ __launch_bounds__(256)
void normalize_k(const float* __restrict__ x, __bf16* __restrict__ out) {
    __shared__ float sb[8];
    int row = blockIdx.x;
    const float* xr = x + (size_t)row * D_;
    float s2 = 0.f;
    for (int i = threadIdx.x; i < D_; i += 256) { float v = xr[i]; s2 += v * v; }
    s2 = block_sum(s2, sb);
    float rn = 1.0f / fmaxf(sqrtf(s2), 1e-12f);
    __bf16* orow = out + (size_t)row * D_;
    for (int i = threadIdx.x; i < D_; i += 256) orow[i] = (__bf16)(xr[i] * rn);
}

__global__ __launch_bounds__(256)
void argmax_k(const float* __restrict__ scores, int* __restrict__ code) {
    __shared__ float sv[8];
    __shared__ int   si[8];
    int row = blockIdx.x;
    const float* sr = scores + (size_t)row * NCODE_;
    float best = -3.4e38f; int bi = 0x7fffffff;
    for (int i = threadIdx.x; i < NCODE_; i += 256) {
        float v = sr[i];
        if (v > best || (v == best && i < bi)) { best = v; bi = i; }
    }
    int lane = threadIdx.x & 31, w = threadIdx.x >> 5;
#pragma unroll
    for (int o = 16; o; o >>= 1) {
        float ov = __shfl_xor(best, o, 32);
        int   oi = __shfl_xor(bi,   o, 32);
        if (ov > best || (ov == best && oi < bi)) { best = ov; bi = oi; }
    }
    if (lane == 0) { sv[w] = best; si[w] = bi; }
    __syncthreads();
    if (threadIdx.x == 0) {
        for (int i = 1; i < 8; i++)
            if (sv[i] > best || (sv[i] == best && si[i] < bi)) { best = sv[i]; bi = si[i]; }
        code[row] = bi;
    }
}

__global__ void lens_k(const int* __restrict__ mel_len, int* __restrict__ out) {
    int b = threadIdx.x;
    if (b < BATCH_) {
        int l = (mel_len[b] + 1) >> 1;
        l = (l + 1) >> 1;
        out[b] = l;
    }
}

// ---------------------------------------------------------------- host
static inline void gemm(hipStream_t st, const __bf16* A, const __bf16* Bw,
                        const float* bias, const float* colsub, const float* residual,
                        float scale, int M, int N, int K, int dogelu,
                        float* outF, __bf16* outH) {
    dim3 g(N / 128, (M + 127) / 128), b(256);
    gemm_bf16_wmma<<<g, b, 0, st>>>(A, Bw, bias, colsub, residual, scale,
                                    M, N, K, dogelu, outF, outH);
}

extern "C" void kernel_launch(void* const* d_in, const int* in_sizes, int n_in,
                              void* d_out, int out_size, void* d_ws, size_t ws_size,
                              hipStream_t stream) {
    const float* mel      = (const float*)d_in[0];
    const int*   mel_len  = (const int*)  d_in[1];
    const float* conv1_w  = (const float*)d_in[2];
    const float* conv1_b  = (const float*)d_in[3];
    // conv2_w = d_in[4] (converted below)
    const float* conv2_b  = (const float*)d_in[5];
    const float* pos_emb  = (const float*)d_in[6];
    const float* ln1_g    = (const float*)d_in[7];
    const float* ln1_b    = (const float*)d_in[8];
    const float* q_w      = (const float*)d_in[9];
    const float* q_b      = (const float*)d_in[10];
    const float* k_w      = (const float*)d_in[11];
    const float* v_w      = (const float*)d_in[12];
    const float* v_b      = (const float*)d_in[13];
    const float* o_w      = (const float*)d_in[14];
    const float* o_b      = (const float*)d_in[15];
    const float* ln2_g    = (const float*)d_in[16];
    const float* ln2_b    = (const float*)d_in[17];
    const float* fc1_w    = (const float*)d_in[18];
    const float* fc1_b    = (const float*)d_in[19];
    const float* fc2_w    = (const float*)d_in[20];
    const float* fc2_b    = (const float*)d_in[21];
    const float* codebook = (const float*)d_in[22];

    char* wsb = (char*)d_ws;
    size_t off = 0;
    auto alloc = [&](size_t bytes) -> void* {
        void* p = wsb + off;
        off += (bytes + 255) & ~(size_t)255;
        return p;
    };

    const size_t DD = (size_t)D_ * D_;
    __bf16* wq  = (__bf16*)alloc(NLAYER_ * DD * 2);
    __bf16* wk  = (__bf16*)alloc(NLAYER_ * DD * 2);
    __bf16* wv  = (__bf16*)alloc(NLAYER_ * DD * 2);
    __bf16* wo  = (__bf16*)alloc(NLAYER_ * DD * 2);
    __bf16* wf1 = (__bf16*)alloc((size_t)NLAYER_ * FF_ * D_ * 2);
    __bf16* wf2 = (__bf16*)alloc((size_t)NLAYER_ * D_ * FF_ * 2);
    __bf16* wc2 = (__bf16*)alloc((size_t)D_ * K2_ * 2);
    __bf16* cb16= (__bf16*)alloc((size_t)NCODE_ * D_ * 2);
    float*  cb2 = (float*) alloc(NCODE_ * 4);
    float*  x1  = (float*) alloc((size_t)BATCH_ * T1_ * D_ * 4);
    // GEMM-A activations padded to MPAD_ rows: async staging may over-read rows
    // [MTOT_, MPAD_), which must stay inside the workspace.
    __bf16* A2  = (__bf16*)alloc((size_t)MPAD_ * K2_ * 2);
    float*  x   = (float*) alloc((size_t)MTOT_ * D_ * 4);
    __bf16* h   = (__bf16*)alloc((size_t)MPAD_ * D_ * 2);
    __bf16* qb  = (__bf16*)alloc((size_t)MTOT_ * D_ * 2);
    __bf16* kb  = (__bf16*)alloc((size_t)MTOT_ * D_ * 2);
    __bf16* vb  = (__bf16*)alloc((size_t)MTOT_ * D_ * 2);
    __bf16* ab  = (__bf16*)alloc((size_t)MPAD_ * D_ * 2);
    __bf16* h1  = (__bf16*)alloc((size_t)MPAD_ * FF_ * 2);
    float*  sc  = (float*) alloc((size_t)MTOT_ * NCODE_ * 4);
    __bf16* xn  = (__bf16*)alloc((size_t)MPAD_ * D_ * 2);

    auto cvt = [&](const float* s, __bf16* d, long n) {
        tobf16_k<<<dim3((unsigned)((n + 255) / 256)), dim3(256), 0, stream>>>(s, d, n);
    };
    cvt(q_w,  wq,  (long)NLAYER_ * DD);
    cvt(k_w,  wk,  (long)NLAYER_ * DD);
    cvt(v_w,  wv,  (long)NLAYER_ * DD);
    cvt(o_w,  wo,  (long)NLAYER_ * DD);
    cvt(fc1_w, wf1, (long)NLAYER_ * FF_ * D_);
    cvt(fc2_w, wf2, (long)NLAYER_ * D_ * FF_);
    cvt((const float*)d_in[4], wc2, (long)D_ * K2_);
    cvt(codebook, cb16, (long)NCODE_ * D_);
    cb2_k<<<NCODE_, 256, 0, stream>>>(codebook, cb2);

    // conv stem
    conv1_k<<<dim3(D_ / 256, T1_, BATCH_), 256, 0, stream>>>(mel, conv1_w, conv1_b, x1);
    im2col_k<<<dim3((unsigned)(((size_t)MTOT_ * K2_ + 255) / 256)), 256, 0, stream>>>(x1, A2);
    gemm(stream, A2, wc2, conv2_b, nullptr, nullptr, 1.0f, MTOT_, D_, K2_, 1, x, nullptr);
    posadd_k<<<dim3((unsigned)((size_t)MTOT_ * D_ / 256)), 256, 0, stream>>>(x, pos_emb);

    const float qkscale = 0.35355339059327373f;  // 64^-0.25
    for (int l = 0; l < NLAYER_; l++) {
        const __bf16* lwq = wq + (size_t)l * DD;
        const __bf16* lwk = wk + (size_t)l * DD;
        const __bf16* lwv = wv + (size_t)l * DD;
        const __bf16* lwo = wo + (size_t)l * DD;
        layernorm_k<<<MTOT_, 256, 0, stream>>>(x, ln1_g + l * D_, ln1_b + l * D_, h);
        gemm(stream, h, lwq, q_b + l * D_, nullptr, nullptr, qkscale, MTOT_, D_, D_, 0, nullptr, qb);
        gemm(stream, h, lwk, nullptr,      nullptr, nullptr, qkscale, MTOT_, D_, D_, 0, nullptr, kb);
        gemm(stream, h, lwv, v_b + l * D_, nullptr, nullptr, 1.0f,    MTOT_, D_, D_, 0, nullptr, vb);
        attention_k<<<dim3((TOK_ + 7) / 8, NHEAD_, BATCH_), 256, 0, stream>>>(qb, kb, vb, mel_len, ab);
        gemm(stream, ab, lwo, o_b + l * D_, nullptr, x, 1.0f, MTOT_, D_, D_, 0, x, nullptr);
        layernorm_k<<<MTOT_, 256, 0, stream>>>(x, ln2_g + l * D_, ln2_b + l * D_, h);
        gemm(stream, h, wf1 + (size_t)l * FF_ * D_, fc1_b + l * FF_, nullptr, nullptr, 1.0f,
             MTOT_, FF_, D_, 1, nullptr, h1);
        gemm(stream, h1, wf2 + (size_t)l * D_ * FF_, fc2_b + l * D_, nullptr, x, 1.0f,
             MTOT_, D_, FF_, 0, x, nullptr);
    }

    // quantize: argmax_n( 2*xn.cb^T - |cb|^2 )   (row-constant |xn|^2 dropped)
    normalize_k<<<MTOT_, 256, 0, stream>>>(x, xn);
    gemm(stream, xn, cb16, nullptr, cb2, nullptr, 2.0f, MTOT_, NCODE_, D_, 0, sc, nullptr);
    int* code = (int*)d_out;
    argmax_k<<<MTOT_, 256, 0, stream>>>(sc, code);
    lens_k<<<1, BATCH_, 0, stream>>>(mel_len, code + MTOT_);
    (void)in_sizes; (void)n_in; (void)out_size; (void)ws_size;
}
